// UpBlock_23794118819866
// MI455X (gfx1250) — matrix operand order repaired and checked by reading, not compiled, not run on hardware
//
#include <hip/hip_runtime.h>

// MI455X / gfx1250, wave32, WMMA f16->f32 path.
// Pipeline: [weight f16 prep] -> k_proj (theta/phi/g) -> k_attn (streaming
// softmax attention, beta never materialized) -> k_mlp (o_w proj + residual +
// 2-layer ReLU MLP). All GEMMs via v_wmma_f32_16x16x32_f16.
//
// All WMMA fragments are assembled from 16B-aligned 8-half vector loads
// (v8h -> global_load_b128 / ds_load_b128) concatenated with shufflevector,
// so the hot loops are pure b128 traffic feeding the WMMA pipe.

typedef __attribute__((ext_vector_type(16))) _Float16 v16h;
typedef __attribute__((ext_vector_type(8)))  _Float16 v8h;
typedef __attribute__((ext_vector_type(8)))  float    v8f;

#define BN 4
#define LN 4096
#define NN 1024

__device__ __forceinline__ v16h cat8(v8h lo, v8h hi) {
  return __builtin_shufflevector(lo, hi, 0, 1, 2, 3, 4, 5, 6, 7,
                                 8, 9, 10, 11, 12, 13, 14, 15);
}
// A-fragment (16-bit A 16x32, ISA 7.12.2): lane half h holds K runs
// [8h..8h+7] and [16+8h..16+8h+7]  -> two aligned v8h loads.
__device__ __forceinline__ v16h loadA(const _Float16* row, int h) {
  v8h lo = *(const v8h*)(row + 8 * h);
  v8h hi = *(const v8h*)(row + 16 + 8 * h);
  return cat8(lo, hi);
}
// B-fragment: same K order as A: lane half h holds K runs
// [8h..8h+7] and [16+8h..16+8h+7] -> two aligned v8h loads.
__device__ __forceinline__ v16h loadB(const _Float16* row, int h) {
  v8h lo = *(const v8h*)(row + 8 * h);
  v8h hi = *(const v8h*)(row + 16 + 8 * h);
  return cat8(lo, hi);
}

__device__ __forceinline__ v8f wmma16(v16h a, v16h b, v8f c) {
  return __builtin_amdgcn_wmma_f32_16x16x32_f16(
      /*neg_a=*/false, a, /*neg_b=*/false, b,
      /*c_mod=*/(short)0, c, /*reuse_a=*/false, /*reuse_b=*/false);
}

// ---------------- weight prep (f32 -> padded f16) ----------------
__global__ void k_prep_qkv(const float* __restrict__ thw, const float* __restrict__ phw,
                           const float* __restrict__ gw, _Float16* __restrict__ dst) {
  int i = blockIdx.x * 256 + threadIdx.x;
  if (i >= 112 * 160) return;
  int row = i / 160, c = i % 160;
  float v = 0.f;
  if (c < 130) {
    if (row < 16)      v = thw[row * 130 + c];
    else if (row < 32) v = phw[(row - 16) * 130 + c];
    else if (row < 97) v = gw[(row - 32) * 130 + c];
  }
  dst[i] = (_Float16)v;
}
__global__ void k_prep_ow(const float* __restrict__ ow, _Float16* __restrict__ dst) {
  int i = blockIdx.x * 256 + threadIdx.x;
  if (i >= 144 * 96) return;
  int row = i / 96, c = i % 96;
  dst[i] = (_Float16)((row < 130 && c < 65) ? ow[row * 65 + c] : 0.f);
}
__global__ void k_prep_w1(const float* __restrict__ w1, _Float16* __restrict__ dst) {
  int i = blockIdx.x * 256 + threadIdx.x;
  if (i >= 256 * 160) return;
  int row = i / 160, c = i % 160;
  dst[i] = (_Float16)((c < 130) ? w1[row * 130 + c] : 0.f);
}
__global__ void k_prep_w2(const float* __restrict__ w2, _Float16* __restrict__ dst) {
  int i = blockIdx.x * 256 + threadIdx.x;
  if (i >= 128 * 256) return;
  dst[i] = (_Float16)w2[i];
}

// ---------------- kernel 1: QKV projections ----------------
// th32/ph32: [B][L][32] (l-major, cols 16..31 zeroed -> memory-resident K pad)
// g16: [B][80][L] (c-major -> contiguous over j for the P*g^T B-fragments;
//                  rows 65..79 are written as zeros via the zero weight rows)
__global__ __launch_bounds__(32)
void k_proj(const float* __restrict__ inp, const _Float16* __restrict__ wqkv,
            _Float16* __restrict__ th32, _Float16* __restrict__ ph32,
            _Float16* __restrict__ g16) {
  const int t = threadIdx.x, m = t & 15, h = t >> 4;
  const int b = blockIdx.x >> 8, lt = blockIdx.x & 255;
  const int l0 = lt * 16, r = l0 >> 10, n0 = l0 & 1023;
  const float gx = (r & 1) ? 0.2f : -0.2f;
  const float gy = (r & 2) ? 0.2f : -0.2f;

  __shared__ alignas(16) _Float16 net[16][160];  // 16 net columns, K pad 130->160
  for (int idx = t; idx < 16 * 160; idx += 32) {
    int lc = idx & 15, c = idx >> 4;
    float v = 0.f;
    if (c < 128)      v = inp[((size_t)b * 128 + c) * NN + n0 + lc];
    else if (c == 128) v = gx;
    else if (c == 129) v = gy;
    net[lc][c] = (_Float16)v;
  }
  __syncthreads();

  v16h bn[5];
#pragma unroll
  for (int ks = 0; ks < 5; ++ks) bn[ks] = loadB(&net[m][ks * 32], h);

  const size_t lcol = (size_t)b * LN + l0 + m;  // this lane's output column l

#pragma unroll
  for (int mt = 0; mt < 7; ++mt) {  // 112 padded rows: 16 th + 16 ph + 80 g
    v8f acc = {};
#pragma unroll
    for (int ks = 0; ks < 5; ++ks)
      acc = wmma16(loadA(wqkv + (mt * 16 + m) * 160 + ks * 32, h), bn[ks], acc);
    // destination is uniform per mt -> no divergent stores
#pragma unroll
    for (int v = 0; v < 8; ++v) {
      int row = mt * 16 + v + 8 * h;
      _Float16 hv = (_Float16)acc[v];
      if (mt == 0)      th32[lcol * 32 + row] = hv;
      else if (mt == 1) ph32[lcol * 32 + (row - 16)] = hv;
      else              g16[((size_t)b * 80 + row - 32) * LN + l0 + m] = hv;
    }
  }
  // zero the K-pad halves of theta/phi rows (cols 16..31, split across h)
  v8h zpad = {};
  *(v8h*)(th32 + lcol * 32 + 16 + 8 * h) = zpad;
  *(v8h*)(ph32 + lcol * 32 + 16 + 8 * h) = zpad;
}

// ---------------- kernel 2: streaming-softmax attention ----------------
// Per wave: one (b, 16-row i-tile). o^T: [B][L][96] f16 (cols 80..95 zero pad).
__global__ __launch_bounds__(32)
void k_attn(const _Float16* __restrict__ th32, const _Float16* __restrict__ ph32,
            const _Float16* __restrict__ g16, _Float16* __restrict__ ot16) {
  const int t = threadIdx.x, m = t & 15, h = t >> 4;
  const int b = blockIdx.x >> 8, it = blockIdx.x & 255;
  const int i0 = it * 16;

  __shared__ alignas(16) _Float16 pl[16][32];  // exp(S) relayout (D -> A frag)

  // A = theta^T for this i-tile (K=16->32 pad lives in memory as zeros)
  const v16h ath = loadA(th32 + ((size_t)b * LN + i0 + m) * 32, h);

  v8f zero = {};
  v8f osum[5];
#pragma unroll
  for (int t5 = 0; t5 < 5; ++t5) osum[t5] = zero;
  v8f ssum = zero;

  for (int j0 = 0; j0 < LN; j0 += 32) {
    const _Float16* php = ph32 + ((size_t)b * LN + j0 + m) * 32;
    __builtin_prefetch(php + 1024, 0, 1);  // next j-tile's phi -> global_prefetch
    v16h bp0 = loadB(php, h);        // B = phi, cols j0..j0+15
    v16h bp1 = loadB(php + 512, h);  //          cols j0+16..j0+31
    v8f s0 = wmma16(ath, bp0, zero);
    v8f s1 = wmma16(ath, bp1, zero);
#pragma unroll
    for (int v = 0; v < 8; ++v) {    // |S| <= ~2: exp without max-shift is safe
      float e0 = __expf(s0[v]);
      float e1 = __expf(s1[v]);
      ssum[v] += e0 + e1;
      pl[v + 8 * h][m] = (_Float16)e0;       // D row = v+8h, col = m
      pl[v + 8 * h][16 + m] = (_Float16)e1;
    }
    __syncthreads();
    v16h ap = loadA(&pl[m][0], h);   // P as A-fragment (K = 32 j's)
#pragma unroll
    for (int t5 = 0; t5 < 5; ++t5) { // O[i, c] += P * g^T, c tiled 5x16 = 80
      const _Float16* gp = g16 + ((size_t)b * 80 + t5 * 16 + m) * LN + j0;
      osum[t5] = wmma16(ap, loadB(gp, h), osum[t5]);
    }
    __syncthreads();
  }

  // row sums: reduce ssum over the 16 column-lanes of each half
#pragma unroll
  for (int off = 1; off < 16; off <<= 1)
#pragma unroll
    for (int v = 0; v < 8; ++v) ssum[v] += __shfl_xor(ssum[v], off, 32);
#pragma unroll
  for (int v = 0; v < 8; ++v) ssum[v] = 1.0f / ssum[v];

#pragma unroll
  for (int t5 = 0; t5 < 5; ++t5)
#pragma unroll
    for (int v = 0; v < 8; ++v)
      ot16[((size_t)b * LN + i0 + v + 8 * h) * 96 + t5 * 16 + m] =
          (_Float16)(osum[t5][v] * ssum[v]);
#pragma unroll
  for (int v = 0; v < 8; ++v)  // zero K-pad cols 80..95
    ot16[((size_t)b * LN + i0 + v + 8 * h) * 96 + 80 + m] = (_Float16)0.f;
}

// ---------------- kernel 3: o_w proj + residual + MLP ----------------
__global__ __launch_bounds__(32)
void k_mlp(const float* __restrict__ inp, const _Float16* __restrict__ ot16,
           const _Float16* __restrict__ ow16, const _Float16* __restrict__ w116,
           const _Float16* __restrict__ w216, const float* __restrict__ gammap,
           const float* __restrict__ b1, const float* __restrict__ b2,
           float* __restrict__ out) {
  const int t = threadIdx.x, m = t & 15, h = t >> 4;
  const int b = blockIdx.x >> 8, lt = blockIdx.x & 255;
  const int l0 = lt * 16, r = l0 >> 10, n0 = l0 & 1023;
  const float gx = (r & 1) ? 0.2f : -0.2f;
  const float gy = (r & 2) ? 0.2f : -0.2f;

  __shared__ alignas(16) float    netc[16][132];  // residual columns (f32)
  __shared__ alignas(16) _Float16 net2[16][160];  // gamma*o2 + net, pad->160
  __shared__ alignas(16) _Float16 hbuf[16][256];  // hidden layer

  for (int idx = t; idx < 16 * 130; idx += 32) {
    int lc = idx & 15, c = idx >> 4;
    float v;
    if (c < 128)      v = inp[((size_t)b * 128 + c) * NN + n0 + lc];
    else if (c == 128) v = gx;
    else               v = gy;
    netc[lc][c] = v;
  }
  __syncthreads();

  const float gamma = gammap[0];
  v8f zero = {};

  // Phase A: net2 = gamma * (o_w @ o) + net   (M=130->144, K=65->96)
  const _Float16* op = ot16 + ((size_t)b * LN + l0 + m) * 96;
  v16h bo[3];
#pragma unroll
  for (int ks = 0; ks < 3; ++ks) bo[ks] = loadB(op + ks * 32, h);
#pragma unroll
  for (int mt = 0; mt < 9; ++mt) {
    v8f acc = zero;
#pragma unroll
    for (int ks = 0; ks < 3; ++ks)
      acc = wmma16(loadA(ow16 + (mt * 16 + m) * 96 + ks * 32, h), bo[ks], acc);
#pragma unroll
    for (int v = 0; v < 8; ++v) {
      int row = mt * 16 + v + 8 * h;
      if (mt < 8)             // rows < 128: always valid, uniform store
        net2[m][row] = (_Float16)(gamma * acc[v] + netc[m][row]);
      else if (row < 130)     // only the last tile has the ragged edge
        net2[m][row] = (_Float16)(gamma * acc[v] + netc[m][row]);
    }
  }
  for (int idx = t; idx < 16 * 30; idx += 32) {  // zero K-pad cols 130..159
    int lc = idx & 15, c = 130 + (idx >> 4);
    net2[lc][c] = (_Float16)0.f;
  }
  __syncthreads();

  // Phase B: h = relu(w1 @ net2 + b1)   (M=256, K=130->160)
  v16h bn2[5];
#pragma unroll
  for (int ks = 0; ks < 5; ++ks) bn2[ks] = loadB(&net2[m][ks * 32], h);
#pragma unroll
  for (int mt = 0; mt < 16; ++mt) {
    v8f acc = zero;
#pragma unroll
    for (int ks = 0; ks < 5; ++ks)
      acc = wmma16(loadA(w116 + (mt * 16 + m) * 160 + ks * 32, h), bn2[ks], acc);
#pragma unroll
    for (int v = 0; v < 8; ++v) {
      int row = mt * 16 + v + 8 * h;
      hbuf[m][row] = (_Float16)fmaxf(acc[v] + b1[row], 0.f);
    }
  }
  __syncthreads();

  // Phase C: out = relu(w2 @ h + b2)   (M=128, K=256)
  v16h bh[8];
#pragma unroll
  for (int ks = 0; ks < 8; ++ks) bh[ks] = loadB(&hbuf[m][ks * 32], h);
#pragma unroll
  for (int mt = 0; mt < 8; ++mt) {
    v8f acc = zero;
#pragma unroll
    for (int ks = 0; ks < 8; ++ks)
      acc = wmma16(loadA(w216 + (mt * 16 + m) * 256 + ks * 32, h), bh[ks], acc);
#pragma unroll
    for (int v = 0; v < 8; ++v) {
      int row = mt * 16 + v + 8 * h;
      out[((size_t)b * 128 + row) * LN + l0 + m] = fmaxf(acc[v] + b2[row], 0.f);
    }
  }
}

// ---------------- launch ----------------
extern "C" void kernel_launch(void* const* d_in, const int* in_sizes, int n_in,
                              void* d_out, int out_size, void* d_ws, size_t ws_size,
                              hipStream_t stream) {
  const float* inp    = (const float*)d_in[0];
  const float* thw    = (const float*)d_in[1];
  const float* phw    = (const float*)d_in[2];
  const float* gw     = (const float*)d_in[3];
  const float* ow     = (const float*)d_in[4];
  const float* gammap = (const float*)d_in[5];
  const float* w1     = (const float*)d_in[6];
  const float* b1     = (const float*)d_in[7];
  const float* w2     = (const float*)d_in[8];
  const float* b2     = (const float*)d_in[9];
  float* out = (float*)d_out;

  // workspace layout (f16 halves), ~8.1 MB total; all sub-arrays 16B-aligned
  _Float16* th32 = (_Float16*)d_ws;
  _Float16* ph32 = th32 + (size_t)BN * LN * 32;
  _Float16* g16  = ph32 + (size_t)BN * LN * 32;
  _Float16* ot16 = g16  + (size_t)BN * 80 * LN;
  _Float16* wqkv = ot16 + (size_t)BN * LN * 96;
  _Float16* ow16 = wqkv + 112 * 160;
  _Float16* w116 = ow16 + 144 * 96;
  _Float16* w216 = w116 + 256 * 160;

  k_prep_qkv<<<(112 * 160 + 255) / 256, 256, 0, stream>>>(thw, phw, gw, wqkv);
  k_prep_ow <<<(144 * 96  + 255) / 256, 256, 0, stream>>>(ow, ow16);
  k_prep_w1 <<<(256 * 160 + 255) / 256, 256, 0, stream>>>(w1, w116);
  k_prep_w2 <<<(128 * 256 + 255) / 256, 256, 0, stream>>>(w2, w216);

  k_proj<<<BN * 256, 32, 0, stream>>>(inp, wqkv, th32, ph32, g16);
  k_attn<<<BN * 256, 32, 0, stream>>>(th32, ph32, g16, ot16);
  k_mlp <<<BN * 256, 32, 0, stream>>>(inp, ot16, ow16, w116, w216, gammap, b1, b2, out);
}
